// Attn_90744069030604
// MI455X (gfx1250) — matrix operand order repaired and checked by reading, not compiled, not run on hardware
//
#include <hip/hip_runtime.h>

// ---------------------------------------------------------------------------
// Types for CDNA5 WMMA (wave32, 16x16x32 bf16 -> f32 accumulate)
// ---------------------------------------------------------------------------
typedef __attribute__((ext_vector_type(16))) __bf16        v16bf;
typedef __attribute__((ext_vector_type(8)))  float         v8f;
typedef __attribute__((ext_vector_type(8)))  unsigned int  v8u;

__device__ __forceinline__ unsigned short f32_to_bf16(float f) {
    unsigned int u = __float_as_uint(f);
    u += 0x7FFFu + ((u >> 16) & 1u);           // round-to-nearest-even
    return (unsigned short)(u >> 16);
}
__device__ __forceinline__ float bf16_to_f32(unsigned short h) {
    return __uint_as_float(((unsigned int)h) << 16);
}

__device__ __forceinline__ v8f wmma_bf16(v16bf a, v16bf b, v8f c) {
    // (neg_a, A, neg_b, B, c_mod, C, reuse_a, reuse_b)
    return __builtin_amdgcn_wmma_f32_16x16x32_bf16(false, a, false, b, (short)0, c,
                                                   false, false);
}

// ---------------------------------------------------------------------------
// gfx1250 async global->LDS DMA (ASYNCcnt-tracked), ISA ch.10 / 15.18.
// Low 32 bits of a generic pointer into __shared__ are the LDS byte offset
// (flat aperture: isLDS checks ADDR[63:32]; LDS addr = ADDR[31:0]).
// ---------------------------------------------------------------------------
__device__ __forceinline__ unsigned lds_off(const void* p) {
    return (unsigned)(unsigned long long)p;
}
__device__ __forceinline__ void async_copy_b128(unsigned lds_byte_off,
                                                const void* gaddr) {
    asm volatile("global_load_async_to_lds_b128 %0, %1, off"
                 :
                 : "v"(lds_byte_off), "v"(gaddr)
                 : "memory");
}
__device__ __forceinline__ void wait_asynccnt0() {
#if __has_builtin(__builtin_amdgcn_s_wait_asynccnt)
    __builtin_amdgcn_s_wait_asynccnt(0);
#else
    asm volatile("s_wait_asynccnt 0x0" ::: "memory");
#endif
}

// A fragment (16x32 bf16, MxK): lane half = lane>>4, m = mBase + (lane&15)
// element e -> k = e + (e&8) + half*8   (ISA 16-bit A layout); pairs contiguous.
__device__ __forceinline__ v16bf load_a_frag(const unsigned short* __restrict__ smem,
                                             int ld, int mBase, int lane) {
    const int half = lane >> 4;
    const unsigned short* row = smem + (mBase + (lane & 15)) * ld + half * 8;
    v8u pack;
#pragma unroll
    for (int j = 0; j < 8; ++j) {
        int e = 2 * j;
        pack[j] = *(const unsigned int*)(row + e + (e & 8));
    }
    return __builtin_bit_cast(v16bf, pack);
}

// B fragment (32x16 bf16, KxN), staged TRANSPOSED in LDS as [n][k]:
// lane n = nBase + (lane&15), element e -> k = e + (e&8) + (lane>>4)*8 (ISA B layout).
__device__ __forceinline__ v16bf load_b_frag(const unsigned short* __restrict__ smem,
                                             int ld, int nBase, int lane) {
    const unsigned short* col = smem + (nBase + (lane & 15)) * ld + (lane >> 4) * 8;
    v8u pack;
#pragma unroll
    for (int j = 0; j < 8; ++j)
        pack[j] = *(const unsigned int*)(col + 2 * j + ((2 * j) & 8));
    return __builtin_bit_cast(v16bf, pack);
}

// ---------------------------------------------------------------------------
// Generic strided-batched bf16 GEMM:  C[z] = A[z] (MxK) * B[z] (KxN)  (+epilogue)
// batch offset = (z/split)*outer + (z%split)*inner  per matrix
// MODE 0: store bf16
// MODE 1: +bias[n] +res[idx], store f32
// MODE 2: +bias[n], exact GELU, store bf16
// ---------------------------------------------------------------------------
struct BatchMap { long long outer; long long inner; int split; };
__device__ __forceinline__ long long boff(BatchMap m, int z) {
    return (long long)(z / m.split) * m.outer + (long long)(z % m.split) * m.inner;
}

#define LDT 40   // LDS row pitch (ushorts): even, 80B rows keep uint4 alignment

union U16x16 { uint4 q[2]; unsigned short s[16]; };

template <int MODE>
__global__ __launch_bounds__(256) void gemm_bf16_kernel(
    const unsigned short* __restrict__ A, const unsigned short* __restrict__ B,
    const float* __restrict__ bias, const float* __restrict__ res,
    float* __restrict__ outF, unsigned short* __restrict__ outBf,
    int M, int N, int K, int lda, int ldb, int ldc,
    BatchMap mA, BatchMap mB, BatchMap mC) {
    __shared__ unsigned short As[128 * LDT];
    __shared__ unsigned short Bs[128 * LDT];

    const int tid  = threadIdx.x;
    const int lane = tid & 31;
    const int wave = tid >> 5;
    const int wm   = wave >> 1;          // 0..3 -> M
    const int wn   = wave & 1;           // 0..1 -> N
    const int z    = blockIdx.z;
    const int m0b  = blockIdx.y * 128;
    const int n0b  = blockIdx.x * 128;

    const long long offA = boff(mA, z);
    const long long offB = boff(mB, z);
    const long long offC = boff(mC, z);

    v8f acc[2][4];
    const v8f zero8 = {};
#pragma unroll
    for (int i = 0; i < 2; ++i)
#pragma unroll
        for (int j = 0; j < 4; ++j) acc[i][j] = zero8;

    // A tile load map: 2 threads/row, 16 ushorts each
    const int ar = tid >> 1;
    const int ac = (tid & 1) * 16;
    // B tile load map: 8 threads/row-of-k, 16 ushorts along n each
    const int bk = tid >> 3;
    const int bn = (tid & 7) * 16;

    const unsigned dstA = lds_off(&As[ar * LDT + ac]);
    const bool aValid = (m0b + ar) < M;
    const unsigned short* srcA =
        aValid ? (A + offA + (long long)(m0b + ar) * lda + ac) : nullptr;

    for (int k0 = 0; k0 < K; k0 += 32) {
        // ---- stage A (128x32): async DMA straight into LDS ----
        if (aValid) {
            async_copy_b128(dstA, srcA + k0);
            async_copy_b128(dstA + 16, srcA + k0 + 8);
            if (k0 + 32 < K)
                __builtin_prefetch((const void*)(srcA + k0 + 32), 0, 1);
        } else {
            uint4* d = reinterpret_cast<uint4*>(&As[ar * LDT + ac]);
            d[0] = make_uint4(0, 0, 0, 0);
            d[1] = make_uint4(0, 0, 0, 0);
        }
        // ---- stage B (32x128), transposed into Bs[n][k] ----
        {
            U16x16 u;
            const int gn = n0b + bn;
            if (gn < N) {
                const uint4* p = reinterpret_cast<const uint4*>(
                    B + offB + (long long)(k0 + bk) * ldb + gn);
                u.q[0] = p[0];
                u.q[1] = p[1];
                if (k0 + 32 < K)
                    __builtin_prefetch(
                        (const void*)(B + offB + (long long)(k0 + 32 + bk) * ldb + gn),
                        0, 1);
            } else {
                u.q[0] = make_uint4(0, 0, 0, 0);
                u.q[1] = make_uint4(0, 0, 0, 0);
            }
#pragma unroll
            for (int i = 0; i < 16; ++i) Bs[(bn + i) * LDT + bk] = u.s[i];
        }
        wait_asynccnt0();                 // A DMA landed in LDS
        __syncthreads();                  // + B stores visible to all waves

        // ---- 8 WMMAs per wave: 32(M) x 64(N) x 32(K) ----
        const int mW = wm * 32;
        const int nW = wn * 64;
        v16bf a0 = load_a_frag(As, LDT, mW, lane);
        v16bf a1 = load_a_frag(As, LDT, mW + 16, lane);
#pragma unroll
        for (int j = 0; j < 4; ++j) {
            v16bf b = load_b_frag(Bs, LDT, nW + 16 * j, lane);
            acc[0][j] = wmma_bf16(a0, b, acc[0][j]);
            acc[1][j] = wmma_bf16(a1, b, acc[1][j]);
        }
        __syncthreads();
    }

    // ---- epilogue ----
#pragma unroll
    for (int i = 0; i < 2; ++i) {
        const int mBase = m0b + wm * 32 + i * 16 + (lane >> 4) * 8;
#pragma unroll
        for (int j = 0; j < 4; ++j) {
            const int n = n0b + wn * 64 + j * 16 + (lane & 15);
            if (n >= N) continue;
            const float bv = (MODE != 0 && bias) ? bias[n] : 0.0f;
#pragma unroll
            for (int r = 0; r < 8; ++r) {
                const int m = mBase + r;
                if (m >= M) continue;
                const long long idx = offC + (long long)m * ldc + n;
                float v = acc[i][j][r];
                if (MODE == 0) {
                    outBf[idx] = f32_to_bf16(v);
                } else if (MODE == 1) {
                    outF[idx] = v + bv + res[idx];
                } else {
                    v += bv;
                    v = 0.5f * v * (1.0f + erff(v * 0.70710678118654752f));
                    outBf[idx] = f32_to_bf16(v);
                }
            }
        }
    }
}

// ---------------------------------------------------------------------------
// LayerNorm (+ optional sinusoidal positional encoding), f32 in -> bf16 out
// ---------------------------------------------------------------------------
__device__ __forceinline__ float wave_sum(float v) {
#pragma unroll
    for (int o = 16; o; o >>= 1) v += __shfl_xor(v, o, 32);
    return v;
}
__device__ __forceinline__ float wave_max(float v) {
#pragma unroll
    for (int o = 16; o; o >>= 1) v = fmaxf(v, __shfl_xor(v, o, 32));
    return v;
}

template <bool WITH_PE>
__global__ __launch_bounds__(256) void layernorm_kernel(
    const float* __restrict__ x, const float* __restrict__ g,
    const float* __restrict__ b, unsigned short* __restrict__ out, int S) {
    const int row = blockIdx.x;              // b*S + s
    const int tid = threadIdx.x;
    const float* xr = x + (long long)row * 1024;

    float v[4], s = 0.f, ss = 0.f;
#pragma unroll
    for (int i = 0; i < 4; ++i) {
        v[i] = xr[tid + i * 256];
        s += v[i];
        ss += v[i] * v[i];
    }
    __shared__ float sh[16];
    float s1 = wave_sum(s), s2 = wave_sum(ss);
    if ((tid & 31) == 0) { sh[tid >> 5] = s1; sh[8 + (tid >> 5)] = s2; }
    __syncthreads();
    s1 = 0.f; s2 = 0.f;
#pragma unroll
    for (int i = 0; i < 8; ++i) { s1 += sh[i]; s2 += sh[8 + i]; }
    const float mean = s1 * (1.0f / 1024.0f);
    const float var  = s2 * (1.0f / 1024.0f) - mean * mean;
    const float rstd = rsqrtf(var + 1e-6f);

    unsigned short* orow = out + (long long)row * 1024;
#pragma unroll
    for (int i = 0; i < 4; ++i) {
        const int c = tid + i * 256;
        float y = (v[i] - mean) * rstd * g[c] + b[c];
        if (WITH_PE) {
            const int sidx = row & (S - 1);
            const float div = __expf((float)(c & ~1) * (-9.210340371976184f / 1024.0f));
            const float ang = (float)sidx * div;
            y += (c & 1) ? __cosf(ang) : __sinf(ang);
        }
        orow[c] = f32_to_bf16(y);
    }
}

// ---------------------------------------------------------------------------
// q softmax over DH=64 (one wave per row), * DH^-0.5; [b,s,h,:]->[b,h,s,:]
// ---------------------------------------------------------------------------
__global__ __launch_bounds__(256) void q_softmax_kernel(
    const unsigned short* __restrict__ qkv, unsigned short* __restrict__ qout) {
    const int tid = threadIdx.x, lane = tid & 31, wave = tid >> 5;
    const long long row = (long long)blockIdx.x * 8 + wave;   // b*S*H + s*H + h
    const long long bs = row >> 4;
    const int h = (int)(row & 15);
    const unsigned short* src = qkv + bs * 3072 + h * 64 + lane * 2;
    float v0 = bf16_to_f32(src[0]);
    float v1 = bf16_to_f32(src[1]);
    const float mx = wave_max(fmaxf(v0, v1));
    v0 = __expf(v0 - mx);
    v1 = __expf(v1 - mx);
    const float inv = 0.125f / wave_sum(v0 + v1);   // 1/sqrt(64) * 1/sum
    const long long bI = bs >> 12;                  // /S (S=4096)
    const long long sI = bs & 4095;
    unsigned short* dst = qout + (((bI * 16 + h) << 12) + sI) * 64 + lane * 2;
    dst[0] = f32_to_bf16(v0 * inv);
    dst[1] = f32_to_bf16(v1 * inv);
}

// ---------------------------------------------------------------------------
// k softmax over S=4096 (one block per column (b,h,d)); writes k^T [b,h,d,s]
// ---------------------------------------------------------------------------
__global__ __launch_bounds__(256) void k_softmax_kernel(
    const unsigned short* __restrict__ qkv, unsigned short* __restrict__ kt) {
    const int col = blockIdx.x;                 // b*1024 + h*64 + d
    const int d = col & 63, h = (col >> 6) & 15, bI = col >> 10;
    const int tid = threadIdx.x;
    const unsigned short* base =
        qkv + (long long)bI * 4096 * 3072 + 1024 + h * 64 + d;

    float v[16], mx = -1e30f;
#pragma unroll
    for (int i = 0; i < 16; ++i) {
        v[i] = bf16_to_f32(base[(long long)(tid + i * 256) * 3072]);
        mx = fmaxf(mx, v[i]);
    }
    __shared__ float sh[16];
    mx = wave_max(mx);
    if ((tid & 31) == 0) sh[tid >> 5] = mx;
    __syncthreads();
    mx = sh[0];
#pragma unroll
    for (int i = 1; i < 8; ++i) mx = fmaxf(mx, sh[i]);

    float s = 0.f;
#pragma unroll
    for (int i = 0; i < 16; ++i) { v[i] = __expf(v[i] - mx); s += v[i]; }
    s = wave_sum(s);
    if ((tid & 31) == 0) sh[8 + (tid >> 5)] = s;
    __syncthreads();
    s = 0.f;
#pragma unroll
    for (int i = 0; i < 8; ++i) s += sh[8 + i];
    const float inv = 1.0f / s;

    unsigned short* dst = kt + (long long)col * 4096 + tid;
#pragma unroll
    for (int i = 0; i < 16; ++i) dst[i * 256] = f32_to_bf16(v[i] * inv);
}

// ---------------------------------------------------------------------------
// f32 -> bf16 cast (weights)
// ---------------------------------------------------------------------------
__global__ __launch_bounds__(256) void cast_f32_bf16_kernel(
    const float* __restrict__ s, unsigned short* __restrict__ d, int n) {
    const int base = blockIdx.x * 1024 + threadIdx.x;
#pragma unroll
    for (int j = 0; j < 4; ++j) {
        const int k = base + j * 256;
        if (k < n) d[k] = f32_to_bf16(s[k]);
    }
}

// ---------------------------------------------------------------------------
// Host launcher
// ---------------------------------------------------------------------------
static inline BatchMap uniform_map(long long stride) {
    return BatchMap{0, stride, 0x40000000};
}

extern "C" void kernel_launch(void* const* d_in, const int* in_sizes, int n_in,
                              void* d_out, int out_size, void* d_ws, size_t ws_size,
                              hipStream_t stream) {
    (void)in_sizes; (void)n_in; (void)out_size; (void)ws_size;
    const float* x     = (const float*)d_in[0];
    const float* ln1_g = (const float*)d_in[1];
    const float* ln1_b = (const float*)d_in[2];
    const float* Wqkv  = (const float*)d_in[3];
    const float* Wo    = (const float*)d_in[4];
    const float* bo    = (const float*)d_in[5];
    const float* ln2_g = (const float*)d_in[6];
    const float* ln2_b = (const float*)d_in[7];
    const float* W1    = (const float*)d_in[8];
    const float* b1    = (const float*)d_in[9];
    const float* W2    = (const float*)d_in[10];
    const float* b2    = (const float*)d_in[11];
    float* out = (float*)d_out;

    const long long ROWS = 16384;   // B*S
    char* ws = (char*)d_ws;
    size_t o = 0;
    auto alloc = [&](size_t bytes) {
        size_t r = o;
        o = (o + bytes + 255) & ~(size_t)255;
        return r;
    };
    unsigned short* wqkv_bf = (unsigned short*)(ws + alloc(1024ll * 3072 * 2));
    unsigned short* wo_bf   = (unsigned short*)(ws + alloc(1024ll * 1024 * 2));
    unsigned short* w1_bf   = (unsigned short*)(ws + alloc(1024ll * 1024 * 2));
    unsigned short* w2_bf   = (unsigned short*)(ws + alloc(1024ll * 1024 * 2));
    size_t h_off   = alloc(ROWS * 1024 * 2);
    size_t qkv_off = alloc(ROWS * 3072 * 2);
    size_t q_off   = alloc(ROWS * 1024 * 2);
    size_t kt_off  = alloc(ROWS * 1024 * 2);
    unsigned short* ctx_bf = (unsigned short*)(ws + alloc(64ll * 64 * 64 * 2));

    unsigned short* h_bf   = (unsigned short*)(ws + h_off);
    unsigned short* qkv_bf = (unsigned short*)(ws + qkv_off);
    unsigned short* q_sm   = (unsigned short*)(ws + q_off);
    unsigned short* kt     = (unsigned short*)(ws + kt_off);
    // region reuse (in dataflow order, each region dead before rebirth):
    unsigned short* attn_bf = kt;                        // after ctx GEMM
    float*          x2      = (float*)(ws + qkv_off);    // after ctx GEMM
    unsigned short* y_bf    = q_sm;                      // after attn GEMM
    unsigned short* t_bf    = h_bf;                      // after QKV GEMM

    const BatchMap one = uniform_map(0);

    // 1) weights -> bf16
    cast_f32_bf16_kernel<<<3072, 256, 0, stream>>>(Wqkv, wqkv_bf, 1024 * 3072);
    cast_f32_bf16_kernel<<<1024, 256, 0, stream>>>(Wo, wo_bf, 1024 * 1024);
    cast_f32_bf16_kernel<<<1024, 256, 0, stream>>>(W1, w1_bf, 1024 * 1024);
    cast_f32_bf16_kernel<<<1024, 256, 0, stream>>>(W2, w2_bf, 1024 * 1024);

    // 2) h = LN1(x) + PE   -> bf16
    layernorm_kernel<true><<<16384, 256, 0, stream>>>(x, ln1_g, ln1_b, h_bf, 4096);

    // 3) qkv = h @ Wqkv    [16384 x 3072]
    gemm_bf16_kernel<0><<<dim3(24, 128, 1), 256, 0, stream>>>(
        h_bf, wqkv_bf, nullptr, nullptr, nullptr, qkv_bf,
        16384, 3072, 1024, 1024, 3072, 3072, one, one, one);

    // 4) q softmax (rows of 64) -> q_sm [b,h,s,d];  k softmax (over S) -> kt [b,h,d,s]
    q_softmax_kernel<<<32768, 256, 0, stream>>>(qkv_bf, q_sm);
    k_softmax_kernel<<<4096, 256, 0, stream>>>(qkv_bf, kt);

    // 5) ctx[b,h] = k^T @ v : 64x64x4096, batch=64; v read in-place from qkv
    gemm_bf16_kernel<0><<<dim3(1, 1, 64), 256, 0, stream>>>(
        kt, qkv_bf + 2048, nullptr, nullptr, nullptr, ctx_bf,
        64, 64, 4096, 4096, 3072, 64,
        uniform_map(64ll * 4096),
        BatchMap{4096ll * 3072, 64, 16},
        uniform_map(64ll * 64));

    // 6) attn[b,h] = q_sm @ ctx : 4096x64x64, batch=64 -> [b,s,h*64+e] (ldc 1024)
    gemm_bf16_kernel<0><<<dim3(1, 32, 64), 256, 0, stream>>>(
        q_sm, ctx_bf, nullptr, nullptr, nullptr, attn_bf,
        4096, 64, 64, 64, 64, 1024,
        uniform_map(4096ll * 64),
        uniform_map(64ll * 64),
        BatchMap{4096ll * 1024, 64, 16});

    // 7) x2 = x + attn @ Wo + bo   (f32)
    gemm_bf16_kernel<1><<<dim3(8, 128, 1), 256, 0, stream>>>(
        attn_bf, wo_bf, bo, x, x2, nullptr,
        16384, 1024, 1024, 1024, 1024, 1024, one, one, one);

    // 8) y = LN2(x2) -> bf16
    layernorm_kernel<false><<<16384, 256, 0, stream>>>(x2, ln2_g, ln2_b, y_bf, 4096);

    // 9) t = GELU(y @ W1 + b1) -> bf16
    gemm_bf16_kernel<2><<<dim3(8, 128, 1), 256, 0, stream>>>(
        y_bf, w1_bf, b1, nullptr, nullptr, t_bf,
        16384, 1024, 1024, 1024, 1024, 1024, one, one, one);

    // 10) out = x2 + t @ W2 + b2   (f32)
    gemm_bf16_kernel<1><<<dim3(8, 128, 1), 256, 0, stream>>>(
        t_bf, w2_bf, b2, x2, out, nullptr,
        16384, 1024, 1024, 1024, 1024, 1024, one, one, one);
}
